// GraphConvolution_59304908423183
// MI455X (gfx1250) — matrix operand order repaired and checked by reading, not compiled, not run on hardware
//
#include <hip/hip_runtime.h>
#include <hip/hip_bf16.h>

// ---------------------------------------------------------------------------
// GraphConvolution (dual-gated edge GEMM + masked BN) for gfx1250 / MI455X.
//   B=32, N=256, L=16, X=256, Y=64, Z=2X+Y=576
// Edge-batched GEMM (16 x 576) @ (576 x 512) per node on
// v_wmma_f32_16x16x32_f16 with f32 accumulation.
// Weights pre-packed f16 in WMMA B-fragment order, laid out [kt][ntg][lane][16]
// so the hot loop reads all 4 fragments of a k-step from ONE pointer with
// constant immediate offsets. Gate epilogue uses gfx1250 hardware
// transcendentals (v_tanh_f32 / v_exp_f32 / v_rcp_f32).
// ---------------------------------------------------------------------------

typedef __attribute__((ext_vector_type(16))) _Float16 v16h;
typedef __attribute__((ext_vector_type(8)))  _Float16 v8h;
typedef __attribute__((ext_vector_type(4)))  _Float16 v4h;
typedef __attribute__((ext_vector_type(8)))  float    v8f;
typedef __attribute__((ext_vector_type(4)))  float    f32x4;

namespace cfg {
constexpr int Bn   = 32;    // batch
constexpr int Nn   = 256;   // nodes
constexpr int Ln   = 16;    // neighbors (= WMMA M)
constexpr int Xn   = 256;   // node feature dim
constexpr int Yn   = 64;    // edge feature dim
constexpr int Zn   = 2 * Xn + Yn;      // 576
constexpr int KT   = Zn / 32;          // 18 k-tiles of 32
constexpr int NTG  = Xn / 16;          // 16 n-tiles per gemm
constexpr int NODES_PER_WG = 8;
constexpr float BN_EPS = 1e-3f;
// packed weight strides, in f16 elements
constexpr int FRAG_ELEMS = 32 * 16;              // 512  : one 32x16 B fragment
constexpr int KT_STRIDE  = 2 * NTG * FRAG_ELEMS; // 16384: one k-step (32 frags)
}

__device__ __forceinline__ float fast_sigmoid(float x) {
    return __builtin_amdgcn_rcpf(1.0f + __expf(-x));
}

__device__ __forceinline__ float fast_tanh(float x) {
#if __has_builtin(__builtin_amdgcn_tanhf)
    return __builtin_amdgcn_tanhf(x);        // gfx1250 v_tanh_f32
#else
    // tanh(x) = 1 - 2/(exp(2x)+1)
    return 1.0f - 2.0f * __builtin_amdgcn_rcpf(__expf(2.0f * x) + 1.0f);
#endif
}

// ---------------------------------------------------------------------------
// Pack wf (Z x X) and ws (Z x X) fp32 weights into f16 WMMA B-fragments:
//   packed[kt][ntg][lane][j], ntg = g*16 + ntile, 16 f16 per lane (one v16h).
// Dense 16-bit B (32x16) layout: lane c<16 holds column c, K=kt*32+j (j<8),
// kt*32+16+(j-8) (j>=8); lane c+16 holds column c, same with K offset +8.
// ---------------------------------------------------------------------------
__global__ void pack_weights_f16(const float* __restrict__ wf,
                                 const float* __restrict__ ws,
                                 _Float16* __restrict__ packed) {
    using namespace cfg;
    const int total = KT * KT_STRIDE;               // 294912
    int idx = blockIdx.x * blockDim.x + threadIdx.x;
    if (idx >= total) return;
    const int j    = idx & 15;
    const int lane = (idx >> 4) & 31;
    const int ntg  = (idx >> 9) & 31;
    const int kt   = idx >> 14;
    const int g    = ntg >> 4;                      // 0 = wf, 1 = ws
    const int nt   = ntg & 15;
    const int col  = nt * 16 + (lane & 15);
    const int k    = kt * 32 + ((lane >> 4) ? 8 : 0) + ((j >> 3) ? 16 : 0) + (j & 7);
    const float* W = g ? ws : wf;
    packed[idx] = (_Float16)W[k * Xn + col];
}

// ---------------------------------------------------------------------------
// Main fused kernel: one workgroup (8 waves) per 8 nodes; 2 nodes in flight.
// Wave w owns x-columns [w*32, w*32+32) for BOTH gemms (f and s), so the
// sigmoid*tanh gate is wave-local.
// ---------------------------------------------------------------------------
__global__ __launch_bounds__(256)
void gated_graph_conv(const float* __restrict__ v,
                      const int*   __restrict__ nl,
                      const float* __restrict__ e,
                      const float* __restrict__ bfp,
                      const float* __restrict__ bsp,
                      const float* __restrict__ gam,
                      const float* __restrict__ bet,
                      const float* __restrict__ mmean,
                      const float* __restrict__ mvar,
                      const _Float16* __restrict__ packedW,
                      float* __restrict__ out) {
    using namespace cfg;
    __shared__ _Float16 zbuf[2][Ln * Zn];      // 36 KB: edge feature tiles (f16)
    __shared__ float    vnext[2][Xn];          // pre-BN outputs
    __shared__ unsigned mbits[2];              // neighbor-valid bitmask per node
    __shared__ int      vflag[2];              // BN row-valid flag per node

    const int tid  = threadIdx.x;
    const int lane = tid & 31;
    const int wave = tid >> 5;
    const int b    = blockIdx.x >> 5;          // 32 groups per batch entry
    const int grp  = blockIdx.x & 31;
    const int nodeBase = grp * NODES_PER_WG;

    // Per-wave B-fragment base: lane offset + this wave's first ntile.
    // All 4 fragments of a k-step sit at compile-time offsets from here.
    const _Float16* bpBase =
        packedW + (size_t)lane * 16 + (size_t)(wave * 2) * FRAG_ELEMS;
    // A-fragment LDS base (f16 elements): row + K-chunk select per the
    // 16-bit A 16x32 layout (lanes 16-31 start at K+8).
    const int rowA  = lane & 15;
    const int aBase = rowA * Zn + (lane >> 4) * 8;

    for (int it = 0; it < NODES_PER_WG / 2; ++it) {
        const int n0 = nodeBase + it * 2;

        // ---- stage z = [v_center | v_neighbor*mask | e] as f16 in LDS ----
        {
            const int rowg = tid >> 3;         // 0..31 : 2 nodes x 16 edges
            const int c8   = tid & 7;          // 8 threads per row
            const int nd   = rowg >> 4;
            const int l    = rowg & 15;
            const int n    = n0 + nd;
            const int nlv  = nl[(b * Nn + n) * Ln + l];
            const float m  = (nlv >= 0) ? 1.0f : 0.0f;
            const int nls  = (nlv >= 0) ? nlv : 0;
            const float* vc = v + (size_t)(b * Nn + n)   * Xn;
            const float* vn = v + (size_t)(b * Nn + nls) * Xn;
            const float* ep = e + ((size_t)(b * Nn + n) * Ln + l) * Yn;
            _Float16* dst = &zbuf[nd][l * Zn];
            #pragma unroll
            for (int i = 0; i < 18; ++i) {     // 18 float4 = 72 cols / thread
                const int col = (c8 * 18 + i) * 4;
                f32x4 src;
                if (col < Xn)            src = *(const f32x4*)(vc + col);
                else if (col < 2 * Xn)   src = *(const f32x4*)(vn + (col - Xn)) * m;
                else                     src = *(const f32x4*)(ep + (col - 2 * Xn));
                *(v4h*)(dst + col) = __builtin_convertvector(src, v4h);
            }
        }
        if (tid < 2) {
            unsigned mb = 0;
            const int n = n0 + tid;
            for (int l = 0; l < Ln; ++l)
                if (nl[(b * Nn + n) * Ln + l] >= 0) mb |= (1u << l);
            mbits[tid] = mb;
            vflag[tid] = 1;
        }
        __syncthreads();

        // ---- WMMA k-loop: acc[node][gemm][ntile], 8 x v8f accumulators ----
        v8f acc[2][2][2] = {};
        #pragma unroll 2
        for (int kt = 0; kt < KT; ++kt) {
            // A fragments for both nodes: two contiguous 16B LDS chunks each.
            union { v16h v; v8h h[2]; } ua, ub;
            const _Float16* p0 = &zbuf[0][aBase + kt * 32];
            const _Float16* p1 = &zbuf[1][aBase + kt * 32];
            ua.h[0] = *(const v8h*)(p0);
            ua.h[1] = *(const v8h*)(p0 + 16);
            ub.h[0] = *(const v8h*)(p1);
            ub.h[1] = *(const v8h*)(p1 + 16);
            const v16h a0 = ua.v, a1 = ub.v;

            // All 4 weight fragments: one pointer, constant offsets.
            const _Float16* bp = bpBase + (size_t)kt * KT_STRIDE;
            const v16h wf0 = *(const v16h*)(bp);                          // f, nt 2w
            const v16h wf1 = *(const v16h*)(bp + FRAG_ELEMS);             // f, nt 2w+1
            const v16h ws0 = *(const v16h*)(bp + NTG * FRAG_ELEMS);       // s, nt 2w
            const v16h ws1 = *(const v16h*)(bp + (NTG + 1) * FRAG_ELEMS); // s, 2w+1

            acc[0][0][0] = __builtin_amdgcn_wmma_f32_16x16x32_f16(
                false, a0, false, wf0, (short)0, acc[0][0][0], false, false);
            acc[0][0][1] = __builtin_amdgcn_wmma_f32_16x16x32_f16(
                false, a0, false, wf1, (short)0, acc[0][0][1], false, false);
            acc[0][1][0] = __builtin_amdgcn_wmma_f32_16x16x32_f16(
                false, a0, false, ws0, (short)0, acc[0][1][0], false, false);
            acc[0][1][1] = __builtin_amdgcn_wmma_f32_16x16x32_f16(
                false, a0, false, ws1, (short)0, acc[0][1][1], false, false);
            acc[1][0][0] = __builtin_amdgcn_wmma_f32_16x16x32_f16(
                false, a1, false, wf0, (short)0, acc[1][0][0], false, false);
            acc[1][0][1] = __builtin_amdgcn_wmma_f32_16x16x32_f16(
                false, a1, false, wf1, (short)0, acc[1][0][1], false, false);
            acc[1][1][0] = __builtin_amdgcn_wmma_f32_16x16x32_f16(
                false, a1, false, ws0, (short)0, acc[1][1][0], false, false);
            acc[1][1][1] = __builtin_amdgcn_wmma_f32_16x16x32_f16(
                false, a1, false, ws1, (short)0, acc[1][1][1], false, false);
        }

        // ---- gate + masked edge-sum + residual ----
        // C layout: lane c, reg r -> M = r + (c<16 ? 0 : 8), N = c%16.
        const int rbase = (lane >> 4) * 8;
        #pragma unroll
        for (int nd = 0; nd < 2; ++nd) {
            const unsigned mb = mbits[nd];
            #pragma unroll
            for (int t = 0; t < 2; ++t) {
                const int x  = wave * 32 + t * 16 + (lane & 15);
                const float bfv = bfp[x];
                const float bsv = bsp[x];
                float sum = 0.0f;
                #pragma unroll
                for (int r = 0; r < 8; ++r) {
                    const float fpre = acc[nd][0][t][r] + bfv;
                    const float spre = acc[nd][1][t][r] + bsv;
                    const float gate = fast_sigmoid(fpre) * fast_tanh(spre);
                    const float m = ((mb >> (rbase + r)) & 1u) ? 1.0f : 0.0f;
                    sum += gate * m;
                }
                sum += __shfl_xor(sum, 16, 32);   // fold M halves (rows 0-7 + 8-15)
                if (lane < 16) {
                    const int n = n0 + nd;
                    vnext[nd][x] = v[(size_t)(b * Nn + n) * Xn + x] + sum;
                }
            }
        }
        __syncthreads();

        // ---- BN validity: row valid iff all |elem| > 1e-5 ----
        if (fabsf(vnext[0][tid]) <= 1e-5f) atomicAnd(&vflag[0], 0);
        if (fabsf(vnext[1][tid]) <= 1e-5f) atomicAnd(&vflag[1], 0);
        __syncthreads();

        // ---- inference BN + write-out (tid == x) ----
        {
            const float scale = gam[tid] * rsqrtf(mvar[tid] + BN_EPS);
            const float shift = bet[tid] - mmean[tid] * scale;
            #pragma unroll
            for (int nd = 0; nd < 2; ++nd) {
                const int n = n0 + nd;
                const float val = vnext[nd][tid];
                out[(size_t)(b * Nn + n) * Xn + tid] =
                    vflag[nd] ? (val * scale + shift) : 0.0f;
            }
        }
        __syncthreads();   // LDS reuse fence for next node pair
    }
}

extern "C" void kernel_launch(void* const* d_in, const int* in_sizes, int n_in,
                              void* d_out, int out_size, void* d_ws, size_t ws_size,
                              hipStream_t stream) {
    using namespace cfg;
    (void)in_sizes; (void)n_in; (void)out_size; (void)ws_size;
    const float* v     = (const float*)d_in[0];
    const int*   nl    = (const int*)  d_in[1];
    const float* e     = (const float*)d_in[2];
    const float* wf    = (const float*)d_in[3];
    const float* bf    = (const float*)d_in[4];
    const float* ws    = (const float*)d_in[5];
    const float* bs    = (const float*)d_in[6];
    const float* gam   = (const float*)d_in[7];
    const float* bet   = (const float*)d_in[8];
    const float* mmean = (const float*)d_in[9];
    const float* mvar  = (const float*)d_in[10];
    float*       out   = (float*)d_out;
    _Float16*    packed = (_Float16*)d_ws;     // 18*16384 f16 = 576 KB

    const int packElems = KT * KT_STRIDE;
    pack_weights_f16<<<(packElems + 255) / 256, 256, 0, stream>>>(wf, ws, packed);

    dim3 grid(Bn * (Nn / NODES_PER_WG));       // 32 * 32 = 1024 workgroups
    gated_graph_conv<<<grid, 256, 0, stream>>>(v, nl, e, bf, bs, gam, bet,
                                               mmean, mvar, packed, out);
}
